// RNNDecoder_22076131902079
// MI455X (gfx1250) — matrix operand order repaired and checked by reading, not compiled, not run on hardware
//
#include <hip/hip_runtime.h>

// ---------------------------------------------------------------------------
// Problem constants (from reference)
// ---------------------------------------------------------------------------
static constexpr int TB    = 32;
static constexpr int TA    = 64;
static constexpr int TT    = 128;   // timesteps
static constexpr int NIN   = 64;    // input / output width
static constexpr int NE    = 256;   // embedding width
static constexpr int NH    = 512;   // hidden width
static constexpr int NL    = 64;    // latent width
static constexpr int ROWS  = TB * TA;        // 2048 independent recurrence rows
static constexpr int MROWS = 16;             // rows per workgroup (one WMMA M tile)
static constexpr int NWG   = ROWS / MROWS;   // 128 persistent workgroups

typedef __bf16 bf16_t;
typedef __attribute__((ext_vector_type(16))) __bf16 v16bf;
typedef __attribute__((ext_vector_type(8)))  __bf16 v8bf;
typedef __attribute__((ext_vector_type(8)))  float  v8f;

// D = A(16x32 bf16) * B(32x16 bf16) + C(16x16 f32), full fp32 accumulate
__device__ __forceinline__ v8f wmma_bf16(v16bf a, v16bf b, v8f c) {
  return __builtin_amdgcn_wmma_f32_16x16x32_bf16(
      /*neg_a=*/false, a, /*neg_b=*/false, b,
      /*c_mod=*/(short)0, c, /*reuse_a=*/false, /*reuse_b=*/false);
}

__device__ __forceinline__ v8f splat8(float x) {
  v8f v = {x, x, x, x, x, x, x, x};
  return v;
}

// A operand: 16x32 tile from row-major bf16 LDS buffer [16][ldk] at column k0.
// ISA A layout: lanes 0-15 (M=lane) hold K = k0+0..7 then k0+16..23;
//               lanes 16-31 (M=lane-16) hold K = k0+8..15 then k0+24..31.
// => per lane: two contiguous 16B chunks -> two ds_load_b128.
__device__ __forceinline__ v16bf lds_load_A(const bf16_t* base, int ldk, int k0, int lane) {
  const int m = lane & 15;
  const int c = lane >> 4;
  const bf16_t* p = base + m * ldk + k0 + 8 * c;
  v8bf lo = *(const v8bf*)(p);
  v8bf hi = *(const v8bf*)(p + 16);
  return __builtin_shufflevector(lo, hi, 0,1,2,3,4,5,6,7,8,9,10,11,12,13,14,15);
}

// B operand: 32x16 tile straight from row-major bf16 weight W[N][K] in global.
// ISA B layout: lane%16 = N, lanes 0-15 hold K = k0+0..7 then k0+16..23, lanes 16-31 hold
// K = k0+8..15 then k0+24..31 -> per lane two 16B chunks -> two global_load_b128.
// Weights stay resident in the 192MB L2; no LDS staging / transpose needed.
__device__ __forceinline__ v16bf glb_load_B(const bf16_t* W, int ldk, int n0, int k0, int lane) {
  const int n = lane & 15;
  const int g = lane >> 4;
  const bf16_t* p = W + (size_t)(n0 + n) * ldk + k0 + 8 * g;
  v8bf lo = *(const v8bf*)(p);
  v8bf hi = *(const v8bf*)(p + 16);
  return __builtin_shufflevector(lo, hi, 0,1,2,3,4,5,6,7,8,9,10,11,12,13,14,15);
}

__device__ __forceinline__ float fast_sigmoid(float x) {
  return 1.0f / (1.0f + __expf(-x));
}
__device__ __forceinline__ float fast_tanh(float x) {
  return 1.0f - 2.0f / (__expf(2.0f * x) + 1.0f);
}

// ---------------------------------------------------------------------------
// fp32 -> bf16 weight conversion (one-time pre-pass into workspace)
// ---------------------------------------------------------------------------
__global__ void cvt_f32_to_bf16(const float* __restrict__ src,
                                bf16_t* __restrict__ dst, int n) {
  int i = blockIdx.x * blockDim.x + threadIdx.x;
  int stride = gridDim.x * blockDim.x;
  for (; i < n; i += stride) dst[i] = (bf16_t)src[i];
}

// ---------------------------------------------------------------------------
// Persistent GRU-decoder kernel: one WG owns 16 rows for all 127 steps.
// 256 threads = 8 waves. Wave w owns h-columns [w*64, w*64+64).
//
// GRU identity used: r and z gates only ever see (gx + gh), so the xe@W_ih
// k-loop accumulates into the SAME WMMA accumulators as the h@W_hh k-loop.
// Only the n gate keeps gxn / ghn separate (needed for r * ghn).
// Biases are folded into the accumulator init (C input of the first WMMA).
// ---------------------------------------------------------------------------
__global__ __launch_bounds__(256) void rnn_decoder_kernel(
    const float* __restrict__ latents,   // [ROWS, NL]
    const float* __restrict__ inputs,    // [ROWS, TT, NIN]
    const float* __restrict__ b_lat,     // [NH]
    const float* __restrict__ b_emb,     // [NE]
    const float* __restrict__ b_out,     // [NIN]
    const float* __restrict__ b_ih,      // [3*NH]
    const float* __restrict__ b_hh,      // [3*NH]
    const bf16_t* __restrict__ Wlat,     // [NH][NL]
    const bf16_t* __restrict__ Wemb,     // [NE][NIN]
    const bf16_t* __restrict__ Wih,      // [3*NH][NE]
    const bf16_t* __restrict__ Whh,      // [3*NH][NH]
    const bf16_t* __restrict__ Wout,     // [NIN][NH]
    float* __restrict__ out)             // [ROWS, TT, NIN]
{
  __shared__ float  h_f32[MROWS * NH];   // fp32 master hidden state  (32KB)
  __shared__ bf16_t h_bf [MROWS * NH];   // bf16 shadow for WMMA A    (16KB)
  __shared__ bf16_t xe_bf[MROWS * NE];   // embedded input, bf16      ( 8KB)
  __shared__ float  x_f32[MROWS * NIN];  // fp32 master residual x    ( 4KB)
  __shared__ bf16_t x_bf [MROWS * NIN];  // bf16 shadow of x          ( 2KB)
  __shared__ bf16_t lat_bf[MROWS * NL];  // latents as WMMA A         ( 2KB)

  const int tid  = threadIdx.x;
  const int w    = tid >> 5;         // wave id 0..7
  const int lane = tid & 31;
  const int r0   = blockIdx.x * MROWS;
  const int nlo  = lane & 15;        // N within tile
  const int mg   = (lane >> 4) * 8;  // accumulator M base for this lane

  // ---- stage latents + x0 into LDS ----
  for (int i = tid; i < MROWS * NIN; i += 256) {
    int m = i >> 6, c = i & 63;
    lat_bf[i] = (bf16_t)latents[(size_t)(r0 + m) * NL + c];
    float xv  = inputs[((size_t)(r0 + m) * TT) * NIN + c];
    x_f32[i]  = xv;
    x_bf[i]   = (bf16_t)xv;
  }
  __syncthreads();

  // ---- h = tanh(latents @ W_lat^T + b_lat); wave w -> 4 column tiles ----
  for (int tt = 0; tt < 4; ++tt) {
    const int n0 = w * 64 + tt * 16;
    const int col = n0 + nlo;
    v8f acc = splat8(b_lat[col]);          // bias folded into C init
#pragma unroll
    for (int k = 0; k < NL; k += 32)
      acc = wmma_bf16(lds_load_A(lat_bf, NL, k, lane),
                      glb_load_B(Wlat, NL, n0, k, lane), acc);
#pragma unroll
    for (int i = 0; i < 8; ++i) {
      float hv = fast_tanh(acc[i]);
      h_f32[(mg + i) * NH + col] = hv;
      h_bf [(mg + i) * NH + col] = (bf16_t)hv;
    }
  }

  // ---- emit t=0 output (x0), float4 coalesced per row ----
  {
    int i4 = tid * 4, m = i4 >> 6, c = i4 & 63;
    *(float4*)&out[((size_t)(r0 + m) * TT) * NIN + c] = *(const float4*)&x_f32[m * NIN + c];
  }
  __syncthreads();

  // ---- time loop: all weight traffic streams from L2 (bf16, ~2.4MB total) ----
  for (int t = 1; t < TT; ++t) {
    // xe = x @ W_emb^T + b_emb : wave w -> 2 column tiles of NE
    for (int tt = 0; tt < 2; ++tt) {
      const int n0 = w * 32 + tt * 16;
      const int col = n0 + nlo;
      v8f acc = splat8(b_emb[col]);
#pragma unroll
      for (int k = 0; k < NIN; k += 32)
        acc = wmma_bf16(lds_load_A(x_bf, NIN, k, lane),
                        glb_load_B(Wemb, NIN, n0, k, lane), acc);
#pragma unroll
      for (int i = 0; i < 8; ++i)
        xe_bf[(mg + i) * NE + col] = (bf16_t)acc[i];
    }
    __syncthreads();

    // gates: wave w owns matched (r,z,n) column triples for its 64 h-columns.
    // ar/az accumulate BOTH gh and gx contributions; anh/anx stay separate.
    for (int tt = 0; tt < 4; ++tt) {
      const int n0 = w * 64 + tt * 16;
      const int col = n0 + nlo;
      v8f ar  = splat8(b_ih[col]        + b_hh[col]);        // gxr+ghr+biases
      v8f az  = splat8(b_ih[col + NH]   + b_hh[col + NH]);   // gxz+ghz+biases
      v8f anh = splat8(b_hh[col + 2*NH]);                    // ghn + b_hhn
      v8f anx = splat8(b_ih[col + 2*NH]);                    // gxn + b_ihn

      for (int k = 0; k < NH; k += 32) {                     // h @ W_hh^T
        v16bf a = lds_load_A(h_bf, NH, k, lane);
        ar  = wmma_bf16(a, glb_load_B(Whh, NH, n0,        k, lane), ar);
        az  = wmma_bf16(a, glb_load_B(Whh, NH, n0 + NH,   k, lane), az);
        anh = wmma_bf16(a, glb_load_B(Whh, NH, n0 + 2*NH, k, lane), anh);
      }
      for (int k = 0; k < NE; k += 32) {                     // xe @ W_ih^T
        v16bf a = lds_load_A(xe_bf, NE, k, lane);
        ar  = wmma_bf16(a, glb_load_B(Wih, NE, n0,        k, lane), ar);
        az  = wmma_bf16(a, glb_load_B(Wih, NE, n0 + NH,   k, lane), az);
        anx = wmma_bf16(a, glb_load_B(Wih, NE, n0 + 2*NH, k, lane), anx);
      }
#pragma unroll
      for (int i = 0; i < 8; ++i) {
        float r = fast_sigmoid(ar[i]);
        float z = fast_sigmoid(az[i]);
        float n = fast_tanh(anx[i] + r * anh[i]);
        const int idx = (mg + i) * NH + col;   // columns are wave-private
        h_f32[idx] = (1.0f - z) * n + z * h_f32[idx];
      }
    }
    __syncthreads();

    // regenerate bf16 shadow of h for the next WMMA A reads
    for (int i = tid; i < MROWS * NH; i += 256) h_bf[i] = (bf16_t)h_f32[i];
    __syncthreads();

    // x += h2 @ W_out^T + b_out : waves 0-3 each take one 16-col tile of NIN
    if (w < 4) {
      const int n0 = w * 16;
      const int col = n0 + nlo;
      v8f acc = splat8(b_out[col]);
      for (int k = 0; k < NH; k += 32)
        acc = wmma_bf16(lds_load_A(h_bf, NH, k, lane),
                        glb_load_B(Wout, NH, n0, k, lane), acc);
#pragma unroll
      for (int i = 0; i < 8; ++i) {
        const int idx = (mg + i) * NIN + col;
        float nv = x_f32[idx] + acc[i];
        x_f32[idx] = nv;
        x_bf[idx]  = (bf16_t)nv;
      }
    }
    __syncthreads();

    // emit output for this timestep: out[row][t][:] = x
    {
      int i4 = tid * 4, m = i4 >> 6, c = i4 & 63;
      *(float4*)&out[((size_t)(r0 + m) * TT + t) * NIN + c] =
          *(const float4*)&x_f32[m * NIN + c];
    }
    // no barrier needed: next phase only reads x_bf/h_bf, already fenced
  }
}

// ---------------------------------------------------------------------------
// Host-side launch
// ---------------------------------------------------------------------------
extern "C" void kernel_launch(void* const* d_in, const int* in_sizes, int n_in,
                              void* d_out, int out_size, void* d_ws, size_t ws_size,
                              hipStream_t stream) {
  const float* latents = (const float*)d_in[0];
  const float* inputs  = (const float*)d_in[1];
  const float* W_lat   = (const float*)d_in[2];
  const float* b_lat   = (const float*)d_in[3];
  const float* W_emb   = (const float*)d_in[4];
  const float* b_emb   = (const float*)d_in[5];
  const float* W_out   = (const float*)d_in[6];
  const float* b_out   = (const float*)d_in[7];
  const float* W_ih    = (const float*)d_in[8];
  const float* b_ih    = (const float*)d_in[9];
  const float* W_hh    = (const float*)d_in[10];
  const float* b_hh    = (const float*)d_in[11];
  float* out = (float*)d_out;

  // bf16 weight workspace layout (all offsets 16B-aligned)
  bf16_t* ws      = (bf16_t*)d_ws;
  bf16_t* Wlat_bf = ws;                      // NH*NL      = 32768
  bf16_t* Wemb_bf = ws + 32768;              // NE*NIN     = 16384
  bf16_t* Wih_bf  = ws + 32768 + 16384;      // 3*NH*NE    = 393216
  bf16_t* Whh_bf  = ws + 442368;             // 3*NH*NH    = 786432
  bf16_t* Wout_bf = ws + 1228800;            // NIN*NH     = 32768

  auto cvt = [&](const float* src, bf16_t* dst, int n) {
    int blocks = (n + 255) / 256;
    if (blocks > 1024) blocks = 1024;
    cvt_f32_to_bf16<<<blocks, 256, 0, stream>>>(src, dst, n);
  };
  cvt(W_lat, Wlat_bf, NH * NL);
  cvt(W_emb, Wemb_bf, NE * NIN);
  cvt(W_ih,  Wih_bf,  3 * NH * NE);
  cvt(W_hh,  Whh_bf,  3 * NH * NH);
  cvt(W_out, Wout_bf, NIN * NH);

  rnn_decoder_kernel<<<NWG, 256, 0, stream>>>(
      latents, inputs, b_lat, b_emb, b_out, b_ih, b_hh,
      Wlat_bf, Wemb_bf, Wih_bf, Whh_bf, Wout_bf, out);
}
